// ClusterProbe_4423816314885
// MI455X (gfx1250) — matrix operand order repaired and checked by reading, not compiled, not run on hardware
//
#include <hip/hip_runtime.h>

typedef __attribute__((ext_vector_type(16))) _Float16 v16h;
typedef __attribute__((ext_vector_type(8)))  float    v8f;

#define C_DIM   768
#define N_CLU   64
#define HW      1024        // 32*32
#define NPOS    8192        // 8 * 1024
#define EPSN    1e-12f

// ws layout (floats):
//   [0]              loss accumulator
//   [8 .. 8+64)      cnorm  = 1/max(||cluster_n||, eps)
//   [128 .. 128+8192) fnorm = 1/max(||feat_p||, eps)
//   then int assign[8192]
#define WS_CNORM  8
#define WS_FNORM  128
#define WS_ASSIGN (128 + NPOS)

// ---------------------------------------------------------------- K1: norms
__global__ void __launch_bounds__(256) cp_norms_kernel(
    const float* __restrict__ code, const float* __restrict__ clusters,
    float* __restrict__ ws) {
  if (blockIdx.x == 0 && threadIdx.x == 0) ws[0] = 0.0f;   // loss accumulator
  int wave = blockIdx.x * (blockDim.x >> 5) + (threadIdx.x >> 5);
  int lane = threadIdx.x & 31;
  if (wave < N_CLU) {
    const float* row = clusters + wave * C_DIM;
    float s = 0.0f;
    for (int c = lane; c < C_DIM; c += 32) { float v = row[c]; s += v * v; }
    #pragma unroll
    for (int o = 16; o >= 1; o >>= 1) s += __shfl_xor(s, o, 32);
    if (lane == 0) ws[WS_CNORM + wave] = 1.0f / fmaxf(sqrtf(s), EPSN);
  } else {
    int p = wave - N_CLU;
    if (p < NPOS) {
      int b = p >> 10, hw = p & 1023;
      const float* f = code + (size_t)b * C_DIM * HW + hw;
      float s = 0.0f;
      for (int c = lane; c < C_DIM; c += 32) { float v = f[(size_t)c * HW]; s += v * v; }
      #pragma unroll
      for (int o = 16; o >= 1; o >>= 1) s += __shfl_xor(s, o, 32);
      if (lane == 0) ws[WS_FNORM + p] = 1.0f / fmaxf(sqrtf(s), EPSN);
    }
  }
}

// ------------------------------------------- K2: WMMA GEMM + fused argmax
// One wave computes sim for 16 spatial positions x all 64 clusters:
//   A (4 tiles, 16x32 f16) = clusters pre-scaled by 1/||c||
//   B (32x16 f16)          = feature columns
//   D = 16x16 f32 per tile; argmax over the 64 rows, loss via atomicAdd.
__global__ void __launch_bounds__(256) cp_gemm_argmax_kernel(
    const float* __restrict__ code, const float* __restrict__ clusters,
    float* __restrict__ ws) {
  const float* cnorm = ws + WS_CNORM;
  const float* fnorm = ws + WS_FNORM;
  int* assign = (int*)(ws + WS_ASSIGN);

  int wave = (blockIdx.x * blockDim.x + threadIdx.x) >> 5;   // 0..511
  int lane = threadIdx.x & 31;
  int h    = lane >> 4;        // lane half (K-range selector)
  int lm   = lane & 15;

  int p0 = wave * 16;
  int pb = p0 + lm;            // this lane's B-matrix column (position)
  int b  = pb >> 10, hw = pb & 1023;
  const float* fcol = code + (size_t)b * C_DIM * HW + hw;    // stride HW over channels

  float cs[4];
  #pragma unroll
  for (int t = 0; t < 4; ++t) cs[t] = cnorm[t * 16 + lm];

  v8f acc[4];
  #pragma unroll
  for (int t = 0; t < 4; ++t)
    #pragma unroll
    for (int v = 0; v < 8; ++v) acc[t][v] = 0.0f;

  for (int k0 = 0; k0 < C_DIM; k0 += 32) {
    // B fragment: lane holds col N=lm, K = k0 + 16h + e   (contiguous k order, same as A)
    v16h bf;
    #pragma unroll
    for (int e = 0; e < 16; ++e)
      bf[e] = (_Float16)fcol[(size_t)(k0 + 16 * h + e) * HW];

    #pragma unroll
    for (int t = 0; t < 4; ++t) {
      // A fragment: lane holds row M=lm, K = k0 + 16h + e
      const float* crow = clusters + (t * 16 + lm) * C_DIM;
      float sc = cs[t];
      v16h af;
      #pragma unroll
      for (int e = 0; e < 8; ++e) af[e]     = (_Float16)(crow[k0 + 16 * h + e] * sc);
      #pragma unroll
      for (int e = 0; e < 8; ++e) af[8 + e] = (_Float16)(crow[k0 + 16 * h + 8 + e] * sc);
      acc[t] = __builtin_amdgcn_wmma_f32_16x16x32_f16(
          false, af, false, bf, (short)0, acc[t], false, false);
    }
  }

  // Per-lane argmax: D layout -> lane holds col N=lm, rows (t*16 + v + 8h)
  float best = -3.4e38f; int bestn = 0;
  #pragma unroll
  for (int t = 0; t < 4; ++t)
    #pragma unroll
    for (int v = 0; v < 8; ++v) {
      float val = acc[t][v];
      int   n   = t * 16 + v + 8 * h;
      if (val > best) { best = val; bestn = n; }
    }
  // merge the two lane halves (rows split across lane and lane^16)
  float ob = __shfl_xor(best, 16, 32);
  int   on = __shfl_xor(bestn, 16, 32);
  if (ob > best || (ob == best && on < bestn)) { best = ob; bestn = on; }

  float sim = (h == 0) ? best * fnorm[pb] : 0.0f;
  if (h == 0) assign[pb] = bestn;
  #pragma unroll
  for (int o = 16; o >= 1; o >>= 1) sim += __shfl_xor(sim, o, 32);
  if (lane == 0) atomicAdd(ws, sim);
}

// ---------------------------- K3: bilinear 16x upsample of one-hot + loss
__global__ void __launch_bounds__(256) cp_upsample_kernel(
    const float* __restrict__ ws, float* __restrict__ out) {
  const int* assign = (const int*)(ws + WS_ASSIGN);
  unsigned gid = blockIdx.x * 256u + threadIdx.x;          // < 2^27
  if (gid == 0) out[0] = -ws[0] * (1.0f / (float)NPOS);    // finalize loss

  int ox = gid & 511;
  int oy = (gid >> 9) & 511;
  int n  = (gid >> 18) & 63;
  int b  = gid >> 24;

  float sx = (ox + 0.5f) * 0.0625f - 0.5f;   // half-pixel centers, scale 1/16
  float sy = (oy + 0.5f) * 0.0625f - 0.5f;
  int x0 = (int)floorf(sx), y0 = (int)floorf(sy);
  float wx = sx - (float)x0, wy = sy - (float)y0;
  int x0c = min(max(x0, 0), 31), x1c = min(max(x0 + 1, 0), 31);
  int y0c = min(max(y0, 0), 31), y1c = min(max(y0 + 1, 0), 31);

  const int* ab = assign + b * 1024;
  float v00 = (ab[y0c * 32 + x0c] == n) ? 1.0f : 0.0f;
  float v01 = (ab[y0c * 32 + x1c] == n) ? 1.0f : 0.0f;
  float v10 = (ab[y1c * 32 + x0c] == n) ? 1.0f : 0.0f;
  float v11 = (ab[y1c * 32 + x1c] == n) ? 1.0f : 0.0f;
  float v = (1.0f - wy) * ((1.0f - wx) * v00 + wx * v01)
          +          wy * ((1.0f - wx) * v10 + wx * v11);
  __builtin_nontemporal_store(v, out + 1 + (size_t)gid);   // 536MB stream: bypass cache
}

// ---------------------------------------------------------------- launch
extern "C" void kernel_launch(void* const* d_in, const int* in_sizes, int n_in,
                              void* d_out, int out_size, void* d_ws, size_t ws_size,
                              hipStream_t stream) {
  const float* code     = (const float*)d_in[0];   // [8,768,32,32]
  const float* clusters = (const float*)d_in[1];   // [64,768]
  float* out = (float*)d_out;                      // [1 + 8*64*512*512 + 8*768*32*32]
  float* ws  = (float*)d_ws;                       // needs ~66 KB

  // K1: 64 cluster-norm waves + 8192 feature-norm waves, 8 waves/block
  cp_norms_kernel<<<(N_CLU + NPOS + 7) / 8, 256, 0, stream>>>(code, clusters, ws);

  // K2: 512 waves (16 positions each), 8 waves/block
  cp_gemm_argmax_kernel<<<64, 256, 0, stream>>>(code, clusters, ws);

  // K3: one thread per resized output element
  const size_t resized = (size_t)8 * 64 * 512 * 512;       // 134217728
  cp_upsample_kernel<<<(unsigned)(resized / 256), 256, 0, stream>>>(ws, out);

  // Third output: code passthrough
  hipMemcpyAsync(out + 1 + resized, code, (size_t)8 * 768 * 1024 * sizeof(float),
                 hipMemcpyDeviceToDevice, stream);
}
